// RecurrentSequential_34772055228903
// MI455X (gfx1250) — matrix operand order repaired and checked by reading, not compiled, not run on hardware
//
#include <hip/hip_runtime.h>

// mLSTM, L=2, T=256, B=256, H=1024.
// bf16 WMMA GEMMs (all weights ~40MB bf16, resident in 192MB L2), f32 state math.
// Per timestep: xcvt (layer-0 input to bf16), then per layer: K1 (m = (A Wmih^T)*(h Wmhh^T))
// and K2 (gates GEMM staged through LDS via async loads + fused cell update).

#define TT 256
#define BB 256
#define HH 1024
#define LLAY 2

typedef __attribute__((ext_vector_type(16))) __bf16 bf16x16;
typedef __attribute__((ext_vector_type(8)))  float  f32x8;

union Frag {
  uint4 u[2];
  unsigned short s[16];
  bf16x16 v;
};

__device__ __forceinline__ unsigned short f2bf(float x) {
  // round-to-nearest-even f32 -> bf16 bits
  unsigned u = __float_as_uint(x);
  u = u + 0x7FFFu + ((u >> 16) & 1u);
  return (unsigned short)(u >> 16);
}

__device__ __forceinline__ float sigm(float x) { return 1.0f / (1.0f + __expf(-x)); }

__device__ __forceinline__ bf16x16 load_pair(const unsigned short* p0, const unsigned short* p1) {
  Frag f;
  f.u[0] = *reinterpret_cast<const uint4*>(p0);
  f.u[1] = *reinterpret_cast<const uint4*>(p1);
  return f.v;
}

// A fragment 16x32 bf16, row-major source (ld = H), base pre-offset per lane to (row, k0):
// lanes 0-15 row=lane, K=[0..15]; lanes 16-31 row=lane-16, K=[16..31].
__device__ __forceinline__ bf16x16 load_a_bf(const unsigned short* base) {
  return load_pair(base, base + 8);
}

// B fragment 32x16 bf16: lane n holds column n; lanes 0-15 K=[0..15], lanes 16-31 K=[16..31]:
// 16 contiguous K elements per lane -> two contiguous 16B reads. Works for global W rows or LDS rows.
__device__ __forceinline__ bf16x16 load_b16(const unsigned short* base) {
  return load_pair(base, base + 8);
}

__device__ __forceinline__ f32x8 wmma_bf16(bf16x16 a, bf16x16 b, f32x8 c) {
  return __builtin_amdgcn_wmma_f32_16x16x32_bf16(false, a, false, b, (short)0, c, false, false);
}

// ---------------------------------------------------------------------------
// K1: m = (A @ Wmih^T) * (h @ Wmhh^T), bf16 out [B,H].
// Block 256 thr = 8 waves, tile 32 rows x 64 cols; wave = one 16x16 tile, two
// accumulators. Grid (B/32, H/64) = (8,16). Direct global loads (weight reuse
// is only 2x per block; WGP$ covers it), near-scope prefetch on the weight stream.
// ---------------------------------------------------------------------------
__global__ __launch_bounds__(256) void k1_mul(
    const unsigned short* __restrict__ aB, const unsigned short* __restrict__ hB,
    const unsigned short* __restrict__ wmih, const unsigned short* __restrict__ wmhh,
    unsigned short* __restrict__ m_out)
{
  const int H = HH;
  const int lane = threadIdx.x & 31;
  const int wave = threadIdx.x >> 5;
  const int rbase = blockIdx.x * 32 + (wave >> 2) * 16;
  const int cbase = blockIdx.y * 64 + (wave & 3) * 16;

  const int ar = lane & 15;
  const int ak = (lane >> 4) * 16;
  const int bn = lane & 15;
  const int bk = (lane >> 4) * 16;

  f32x8 accx = {};
  f32x8 acch = {};

  const unsigned short* xA = aB + (size_t)(rbase + ar) * H + ak;
  const unsigned short* hA = hB + (size_t)(rbase + ar) * H + ak;
  const unsigned short* bx = wmih + (size_t)(cbase + bn) * H + bk;
  const unsigned short* bh = wmhh + (size_t)(cbase + bn) * H + bk;

#pragma unroll 2
  for (int k = 0; k < H; k += 32) {
    __builtin_prefetch(bx + k + 256, 0, 3);
    __builtin_prefetch(bh + k + 256, 0, 3);
    bf16x16 a_x = load_a_bf(xA + k);
    bf16x16 a_h = load_a_bf(hA + k);
    accx = wmma_bf16(a_x, load_b16(bx + k), accx);
    acch = wmma_bf16(a_h, load_b16(bh + k), acch);
  }

  // C/D layout: VGPR p -> M = p (+8 for lanes 16-31), N = lane&15.
  const int col = cbase + (lane & 15);
  const int rlo = rbase + ((lane >> 4) << 3);
#pragma unroll
  for (int p = 0; p < 8; ++p) {
    m_out[(size_t)(rlo + p) * H + col] = f2bf(accx[p] * acch[p]);
  }
}

// ---------------------------------------------------------------------------
// K2: gates = A @ Wih^T + m @ Whh^T + bias; fused cell update + mask blending.
// Block 256 thr = 8 waves; tile 64 rows x 32 gate-cols; each wave holds all 4
// gate accumulators for one 16x16 (row,col) tile. Grid (B/64, H/32) = (4,32).
// Weight tiles (128 rows x 64 K, 16KB) are staged into LDS with
// global_load_async_to_lds_b128 (ASYNCcnt), double-buffered, 72-element row
// stride to spread LDS banks; fragments come from ds_load_b128.
// ---------------------------------------------------------------------------
#define K2_STRIDE 72
#define K2_BUFELEMS (128 * K2_STRIDE)

__global__ __launch_bounds__(256) void k2_gates(
    const unsigned short* __restrict__ aB,   // [B,H] bf16 layer input
    const unsigned short* __restrict__ mB,   // [B,H] bf16 m
    const unsigned short* __restrict__ wih, const unsigned short* __restrict__ whh,
    const float* __restrict__ bsum, const int* __restrict__ mask_t,
    float* __restrict__ h_state, float* __restrict__ c_state,
    unsigned short* __restrict__ h_state_bf,
    unsigned short* __restrict__ a_next,     // bf16 unmasked h_new (next-layer input) or null
    float* __restrict__ out_t,               // output slice (last layer) or null
    const float* __restrict__ out_prev, int first_step)
{
  const int H = HH;
  __shared__ unsigned short btile[2][K2_BUFELEMS];

  const int tid  = threadIdx.x;
  const int lane = tid & 31;
  const int wave = tid >> 5;
  const int cb0   = blockIdx.y * 32;
  const int rbase = blockIdx.x * 64 + (wave >> 1) * 16;
  const int cbase = cb0 + (wave & 1) * 16;

  // cooperative staging: thread (srow, half) loads 64B of W row srow (srow = gate*32 + col)
  const int half = tid & 1;
  const int srow = tid >> 1;                         // 0..127
  const int ldsrow = srow * K2_STRIDE + half * 32;   // element offset in LDS row
  const size_t wrow = ((size_t)(srow >> 5) * H + cb0 + (srow & 31)) * H + half * 32;

  // compute-side A bases (per-lane)
  const int ar = lane & 15;
  const int ak = (lane >> 4) * 16;
  const unsigned short* aP1 = aB + (size_t)(rbase + ar) * H + ak;
  const unsigned short* aP2 = mB + (size_t)(rbase + ar) * H + ak;

  // B-fragment LDS element offsets for the 4 gates
  const int bn = lane & 15;
  const int bk = (lane >> 4) * 16;
  const int bfo0 = (0 * 32 + (wave & 1) * 16 + bn) * K2_STRIDE + bk;
  const int bfo1 = (1 * 32 + (wave & 1) * 16 + bn) * K2_STRIDE + bk;
  const int bfo2 = (2 * 32 + (wave & 1) * 16 + bn) * K2_STRIDE + bk;
  const int bfo3 = (3 * 32 + (wave & 1) * 16 + bn) * K2_STRIDE + bk;

  f32x8 acc[4] = {};

  for (int phase = 0; phase < 2; ++phase) {
    const unsigned short* wsrc = (phase ? whh : wih) + wrow;
    const unsigned short* aA   = phase ? aP2 : aP1;

    // prologue: stage 0 -> buffer 0
#pragma unroll
    for (int i = 0; i < 4; ++i) {
      unsigned l = (unsigned)(size_t)&btile[0][ldsrow + i * 8];
      unsigned long long g = (unsigned long long)(size_t)(wsrc + i * 8);
      asm volatile("global_load_async_to_lds_b128 %0, %1, off" :: "v"(l), "v"(g) : "memory");
    }

    for (int s = 0; s < 16; ++s) {
      const int k = s * 64;
      if (s < 15) {
#pragma unroll
        for (int i = 0; i < 4; ++i) {
          unsigned l = (unsigned)(size_t)&btile[(s + 1) & 1][ldsrow + i * 8];
          unsigned long long g = (unsigned long long)(size_t)(wsrc + k + 64 + i * 8);
          asm volatile("global_load_async_to_lds_b128 %0, %1, off" :: "v"(l), "v"(g) : "memory");
        }
        asm volatile("s_wait_asynccnt 0x4" ::: "memory");  // stage s fully landed
      } else {
        asm volatile("s_wait_asynccnt 0x0" ::: "memory");
      }
      __syncthreads();

      const unsigned short* sb = &btile[s & 1][0];
#pragma unroll
      for (int j = 0; j < 2; ++j) {
        bf16x16 a = load_a_bf(aA + k + j * 32);
        acc[0] = wmma_bf16(a, load_b16(sb + bfo0 + j * 32), acc[0]);
        acc[1] = wmma_bf16(a, load_b16(sb + bfo1 + j * 32), acc[1]);
        acc[2] = wmma_bf16(a, load_b16(sb + bfo2 + j * 32), acc[2]);
        acc[3] = wmma_bf16(a, load_b16(sb + bfo3 + j * 32), acc[3]);
      }
      __syncthreads();  // everyone done reading buf[s&1] before it is overwritten
    }
  }

  // Pointwise cell update + mask blending (all f32, in-register: wave owns i,f,g,o).
  const int col = cbase + (lane & 15);
  const int rlo = rbase + ((lane >> 4) << 3);
  const float bi  = bsum[0 * H + col];
  const float bff = bsum[1 * H + col];
  const float bg  = bsum[2 * H + col];
  const float bo  = bsum[3 * H + col];
#pragma unroll
  for (int p = 0; p < 8; ++p) {
    const int b = rlo + p;
    const size_t idx = (size_t)b * H + col;
    const float mt = (float)mask_t[b];
    const float co = c_state[idx];
    const float ig = sigm(acc[0][p] + bi);
    const float fg = sigm(acc[1][p] + bff);
    const float gg = tanhf(acc[2][p] + bg);
    const float og = sigm(acc[3][p] + bo);
    const float cn = fg * co + ig * gg;
    const float hn = og * tanhf(cn);
    const float ho = h_state[idx];
    const float hm = hn * mt + ho * (1.0f - mt);
    const float cm = cn * mt + co * (1.0f - mt);
    c_state[idx] = cm;
    h_state[idx] = hm;
    h_state_bf[idx] = f2bf(hm);
    if (a_next) a_next[idx] = f2bf(hn);            // unmasked h feeds the next layer
    if (out_t)  out_t[idx] = first_step ? hn : (hn * mt + out_prev[idx] * (1.0f - mt));
  }
}

// ------------------------- setup utility kernels ---------------------------
__global__ void f32_to_bf16_k(const float* __restrict__ in, unsigned short* __restrict__ out, int n) {
  int i = blockIdx.x * blockDim.x + threadIdx.x;
  int stride = gridDim.x * blockDim.x;
  for (; i < n; i += stride) out[i] = f2bf(in[i]);
}
__global__ void copy_f32_k(const float* __restrict__ in, float* __restrict__ out, int n) {
  int i = blockIdx.x * blockDim.x + threadIdx.x;
  int stride = gridDim.x * blockDim.x;
  for (; i < n; i += stride) out[i] = in[i];
}
__global__ void add_f32_k(const float* __restrict__ a, const float* __restrict__ b,
                          float* __restrict__ out, int n) {
  int i = blockIdx.x * blockDim.x + threadIdx.x;
  int stride = gridDim.x * blockDim.x;
  for (; i < n; i += stride) out[i] = a[i] + b[i];
}

// ---------------------------------------------------------------------------
extern "C" void kernel_launch(void* const* d_in, const int* in_sizes, int n_in,
                              void* d_out, int out_size, void* d_ws, size_t ws_size,
                              hipStream_t stream) {
  (void)in_sizes; (void)n_in; (void)out_size; (void)ws_size;
  const int T = TT, B = BB, H = HH, L = LLAY;

  const float* x      = (const float*)d_in[0];   // [T,B,H]
  const float* h0     = (const float*)d_in[1];   // [L,B,H]
  const float* c0     = (const float*)d_in[2];   // [L,B,H]
  const int*   mask   = (const int*)d_in[3];     // [T,B]
  const float* w_ih   = (const float*)d_in[4];   // [L,4H,H]
  const float* w_hh   = (const float*)d_in[5];   // [L,4H,H]
  const float* w_mih  = (const float*)d_in[6];   // [L,H,H]
  const float* w_mhh  = (const float*)d_in[7];   // [L,H,H]
  const float* b_ih   = (const float*)d_in[8];   // [L,4H]
  const float* b_hh   = (const float*)d_in[9];   // [L,4H]

  float* out = (float*)d_out;                    // outputs [T,B,H] ++ h_f [L,B,H] ++ c_f [L,B,H]
  float* hF  = out + (size_t)T * B * H;          // persistent h state (== h_f at the end)
  float* cF  = hF + (size_t)L * B * H;           // persistent c state (== c_f at the end)

  // workspace carve-out (~43 MB)
  char* wsb = (char*)d_ws;
  size_t off = 0;
  auto alloc = [&](size_t bytes) -> void* {
    off = (off + 255) & ~(size_t)255;
    void* p = wsb + off;
    off += bytes;
    return p;
  };
  const size_t nWih = (size_t)L * 4 * H * H;
  const size_t nWm  = (size_t)L * H * H;
  const size_t nS   = (size_t)B * H;
  unsigned short* wih_bf  = (unsigned short*)alloc(nWih * 2);
  unsigned short* whh_bf  = (unsigned short*)alloc(nWih * 2);
  unsigned short* wmih_bf = (unsigned short*)alloc(nWm * 2);
  unsigned short* wmhh_bf = (unsigned short*)alloc(nWm * 2);
  unsigned short* h_bf    = (unsigned short*)alloc((size_t)L * nS * 2);
  unsigned short* m_bf    = (unsigned short*)alloc(nS * 2);
  unsigned short* a_bf    = (unsigned short*)alloc(nS * 2);
  unsigned short* x_bf    = (unsigned short*)alloc(nS * 2);
  float*          bsum    = (float*)alloc((size_t)L * 4 * H * 4);

  // per-call setup: bf16 weights, summed bias, initial states
  f32_to_bf16_k<<<2048, 256, 0, stream>>>(w_ih,  wih_bf,  (int)nWih);
  f32_to_bf16_k<<<2048, 256, 0, stream>>>(w_hh,  whh_bf,  (int)nWih);
  f32_to_bf16_k<<<1024, 256, 0, stream>>>(w_mih, wmih_bf, (int)nWm);
  f32_to_bf16_k<<<1024, 256, 0, stream>>>(w_mhh, wmhh_bf, (int)nWm);
  f32_to_bf16_k<<<256,  256, 0, stream>>>(h0, h_bf, (int)(L * nS));
  add_f32_k<<<32, 256, 0, stream>>>(b_ih, b_hh, bsum, L * 4 * H);
  copy_f32_k<<<256, 256, 0, stream>>>(h0, hF, (int)(L * nS));
  copy_f32_k<<<256, 256, 0, stream>>>(c0, cF, (int)(L * nS));

  const dim3 blk(256);
  const dim3 g1(B / 32, H / 64);   // (8,16)
  const dim3 g2(B / 64, H / 32);   // (4,32)

  for (int t = 0; t < T; ++t) {
    const float* xt = x + (size_t)t * B * H;
    const int*   mt = mask + (size_t)t * B;
    // layer-0 input to bf16 (keeps GEMM inner loops conversion-free)
    f32_to_bf16_k<<<128, 256, 0, stream>>>(xt, x_bf, (int)nS);
    // ---- layer 0 ----
    k1_mul<<<g1, blk, 0, stream>>>(x_bf, h_bf, wmih_bf, wmhh_bf, m_bf);
    k2_gates<<<g2, blk, 0, stream>>>(x_bf, m_bf, wih_bf, whh_bf,
        bsum, mt, hF, cF, h_bf, a_bf, nullptr, nullptr, t == 0 ? 1 : 0);
    // ---- layer 1 (input = unmasked h_new of layer 0) ----
    k1_mul<<<g1, blk, 0, stream>>>(a_bf, h_bf + nS,
        wmih_bf + (size_t)H * H, wmhh_bf + (size_t)H * H, m_bf);
    k2_gates<<<g2, blk, 0, stream>>>(a_bf, m_bf,
        wih_bf + (size_t)4 * H * H, whh_bf + (size_t)4 * H * H,
        bsum + 4 * H, mt, hF + nS, cF + nS, h_bf + nS, nullptr,
        out + (size_t)t * B * H,
        t ? out + (size_t)(t - 1) * B * H : out, t == 0 ? 1 : 0);
  }
}
